// External_attention_44976897524182
// MI455X (gfx1250) — matrix operand order repaired and checked by reading, not compiled, not run on hardware
//
#include <hip/hip_runtime.h>
#include <hip/hip_bf16.h>

// ---------------------------------------------------------------------------
// External attention, algebraically folded:
//   W1 = lin0_w @ conv1_w (64x128),  b1 = lin0_w @ conv1_b (64)
//   logits(b,64,n) = W1 @ x + b1
//   s = softmax_n(logits);  attn = s / (1e-9 + sum_k s)
//   W2 = conv2_w @ lin1_w (128x64)
//   out = relu( (W2 @ attn) * bn_scale + bn_shift + x )
// 3 data passes over x (134MB each) + 1 write => ~536MB => ~23us @ 23.3TB/s.
// GEMMs done with v_wmma_f32_16x16x32_bf16 to keep the kernel memory-bound.
// ---------------------------------------------------------------------------

typedef __bf16 bf16;
typedef bf16  v16bf __attribute__((ext_vector_type(16)));
typedef float v8f   __attribute__((ext_vector_type(8)));

#define CCH   128          // channels
#define KEXT  64           // external-attention K
#define NTOT  131072       // d*h*w = 32*64*64
#define NTILE 128          // n-columns per workgroup
#define NBLK  (NTOT / NTILE)   // 1024 blocks per batch
#define XSTR  136          // ldsX row stride (bf16), padded: 136*2B/4 = 68 dwords -> 4 banks/lane
#define LSTR  68           // ldsL row stride (f32), padded
#define BSTR  72           // ldsB row stride (bf16), padded: 144B/4 = 36 dwords -> conflict-free

// ---------------------------------------------------------------------------
// Kernel 0: fold weights, pre-pack A-operands into WMMA VGPR layout.
// A-operand k order used here (contiguous, B matches): lane<16 holds row M=lane with
//   element i -> K = i;  lane>=16 holds row M=lane-16 with K = 16+i.
// We store packedA[(mt*KSTEPS+ks)*32 + lane][16] so each lane does one
// contiguous 32B global load per WMMA.
// ---------------------------------------------------------------------------
__global__ __launch_bounds__(256) void ea_prep(
    const float* __restrict__ conv1_w, const float* __restrict__ conv1_b,
    const float* __restrict__ lin0_w,  const float* __restrict__ lin1_w,
    const float* __restrict__ conv2_w, const float* __restrict__ bn_g,
    const float* __restrict__ bn_b,    const float* __restrict__ bn_m,
    const float* __restrict__ bn_v,
    bf16* __restrict__ pA1, bf16* __restrict__ pA2,
    float* __restrict__ b1, float* __restrict__ scale, float* __restrict__ shift)
{
  const int tid = threadIdx.x;
  // W1 = lin0_w(64x128) @ conv1_w(128x128), packed: mt in [0,4), ks in [0,4)
  for (int idx = tid; idx < KEXT * CCH; idx += 256) {
    int k = idx >> 7, c = idx & 127;
    float acc = 0.f;
    for (int j = 0; j < CCH; ++j) acc += lin0_w[k * CCH + j] * conv1_w[j * CCH + c];
    int mt = k >> 4, ks = c >> 5, kk = c & 31;
    int lane = (k & 15) + (((kk >> 4) & 1) << 4);
    int i    = kk & 15;
    pA1[(((mt * 4 + ks) * 32) + lane) * 16 + i] = (bf16)acc;
  }
  // W2 = conv2_w(128x128) @ lin1_w(128x64), packed: mt in [0,8), ks in [0,2)
  for (int idx = tid; idx < CCH * KEXT; idx += 256) {
    int o = idx >> 6, k = idx & 63;
    float acc = 0.f;
    for (int j = 0; j < CCH; ++j) acc += conv2_w[o * CCH + j] * lin1_w[j * KEXT + k];
    int mt = o >> 4, ks = k >> 5, kk = k & 31;
    int lane = (o & 15) + (((kk >> 4) & 1) << 4);
    int i    = kk & 15;
    pA2[(((mt * 2 + ks) * 32) + lane) * 16 + i] = (bf16)acc;
  }
  if (tid < KEXT) {
    float acc = 0.f;
    for (int j = 0; j < CCH; ++j) acc += lin0_w[tid * CCH + j] * conv1_b[j];
    b1[tid] = acc;
  }
  if (tid < CCH) {
    float inv = rsqrtf(bn_v[tid] + 1e-5f);
    scale[tid] = bn_g[tid] * inv;
    shift[tid] = bn_b[tid] - bn_m[tid] * bn_g[tid] * inv;
  }
}

// ---------------------------------------------------------------------------
// Shared WMMA helper: compute logits tile (64 k x NTILE n) from ldsX into
// ldsL[n][k] (f32, bias added). Executed by all 8 waves of a 256-thread block.
// ---------------------------------------------------------------------------
__device__ __forceinline__ void wmma_logits(const bf16* __restrict__ pA1,
                                            const bf16* ldsX, float* ldsL,
                                            const float* sb1,
                                            int wave, int lane)
{
  const int halfsel = lane >> 4;
  const int nn = wave * 16 + (lane & 15);     // each wave owns one 16-col n-subtile
  for (int mt = 0; mt < 4; ++mt) {
    v8f acc = {};
    #pragma unroll
    for (int ks = 0; ks < 4; ++ks) {
      v16bf a = *(const v16bf*)(pA1 + (((mt * 4 + ks) * 32 + lane) * 16));
      v16bf b = *(const v16bf*)(ldsX + nn * XSTR + ks * 32 + halfsel * 16);
      acc = __builtin_amdgcn_wmma_f32_16x16x32_bf16(false, a, false, b,
                                                    (short)0, acc, false, false);
    }
    const int kbase = mt * 16 + halfsel * 8;     // D: vgpr r -> M = r + 8*halfsel
    float* op = ldsL + nn * LSTR + kbase;        // contiguous 32B store per lane
    #pragma unroll
    for (int r = 0; r < 8; ++r) op[r] = acc[r] + sb1[kbase + r];
  }
}

// ---------------------------------------------------------------------------
// Pass 1: per-tile logits -> per-block (rowmax, sum exp(l - rowmax)) partials.
// ---------------------------------------------------------------------------
__global__ __launch_bounds__(256) void ea_pass1(
    const float* __restrict__ x, const bf16* __restrict__ pA1,
    const float* __restrict__ b1, float* __restrict__ partials)
{
  __shared__ __align__(16) bf16  ldsX[NTILE * XSTR];
  __shared__ __align__(16) float ldsL[NTILE * LSTR];
  __shared__ float sb1[KEXT];

  const int tid   = threadIdx.x;
  const int n0    = blockIdx.x * NTILE;
  const int batch = blockIdx.y;
  const float* xb = x + (size_t)batch * CCH * NTOT;

  if (tid < KEXT) sb1[tid] = b1[tid];
  for (int e = tid; e < CCH * NTILE; e += 256) {     // coalesced: nn contiguous
    int c = e >> 7, nn = e & 127;
    ldsX[nn * XSTR + c] = (bf16)xb[(size_t)c * NTOT + n0 + nn];
  }
  __syncthreads();

  wmma_logits(pA1, ldsX, ldsL, sb1, tid >> 5, tid & 31);
  __syncthreads();

  if (tid < KEXT) {                                  // per-k reduction over tile
    float m = -3.4e38f;
    for (int j = 0; j < NTILE; ++j) m = fmaxf(m, ldsL[j * LSTR + tid]);
    float s = 0.f;
    for (int j = 0; j < NTILE; ++j) s += __expf(ldsL[j * LSTR + tid] - m);
    size_t pi = (((size_t)batch * NBLK + blockIdx.x) * KEXT + tid) * 2;
    partials[pi] = m;
    partials[pi + 1] = s;
  }
}

// ---------------------------------------------------------------------------
// Pass 2: combine per-block partials (online-softmax merge) -> rowmax, 1/rowsum.
// ---------------------------------------------------------------------------
__global__ __launch_bounds__(128) void ea_pass2(
    const float* __restrict__ partials,
    float* __restrict__ rowmax, float* __restrict__ rowinv)
{
  const int row = threadIdx.x;                 // 0..127 = b*64 rows
  const int batch = row >> 6, k = row & 63;
  float M = -3.4e38f;
  for (int blk = 0; blk < NBLK; ++blk)
    M = fmaxf(M, partials[(((size_t)batch * NBLK + blk) * KEXT + k) * 2]);
  float S = 0.f;
  for (int blk = 0; blk < NBLK; ++blk) {
    size_t pi = (((size_t)batch * NBLK + blk) * KEXT + k) * 2;
    S += partials[pi + 1] * __expf(partials[pi] - M);
  }
  rowmax[row] = M;
  rowinv[row] = 1.f / S;
}

// ---------------------------------------------------------------------------
// Pass 3: recompute logits, softmax + L1-renorm over k, z = W2 @ attn,
// BN affine + residual + relu, store.
// ---------------------------------------------------------------------------
__global__ __launch_bounds__(256) void ea_pass3(
    const float* __restrict__ x, const bf16* __restrict__ pA1,
    const bf16* __restrict__ pA2, const float* __restrict__ b1,
    const float* __restrict__ scale, const float* __restrict__ shift,
    const float* __restrict__ rowmax, const float* __restrict__ rowinv,
    float* __restrict__ out)
{
  __shared__ __align__(16) bf16  ldsX[NTILE * XSTR];
  __shared__ __align__(16) float ldsL[NTILE * LSTR];
  __shared__ __align__(16) bf16  ldsB[NTILE * BSTR];
  __shared__ float sb1[KEXT], srmax[KEXT], srinv[KEXT], sscale[CCH], sshift[CCH];

  const int tid   = threadIdx.x;
  const int n0    = blockIdx.x * NTILE;
  const int batch = blockIdx.y;
  const float* xb = x + (size_t)batch * CCH * NTOT;
  float* ob       = out + (size_t)batch * CCH * NTOT;

  if (tid < KEXT) {
    sb1[tid]   = b1[tid];
    srmax[tid] = rowmax[batch * KEXT + tid];
    srinv[tid] = rowinv[batch * KEXT + tid];
  }
  if (tid < CCH) { sscale[tid] = scale[tid]; sshift[tid] = shift[tid]; }
  for (int e = tid; e < CCH * NTILE; e += 256) {
    int c = e >> 7, nn = e & 127;
    ldsX[nn * XSTR + c] = (bf16)xb[(size_t)c * NTOT + n0 + nn];
  }
  __syncthreads();

  const int wave = tid >> 5, lane = tid & 31, halfsel = lane >> 4;
  wmma_logits(pA1, ldsX, ldsL, sb1, wave, lane);
  __syncthreads();

  // softmax over n (precomputed stats) + L1 renorm over k, per column.
  if (tid < NTILE) {
    const int nn = tid;
    float colsum = 0.f;
    #pragma unroll 4
    for (int k = 0; k < KEXT; ++k) {
      float s = __expf(ldsL[nn * LSTR + k] - srmax[k]) * srinv[k];
      ldsL[nn * LSTR + k] = s;
      colsum += s;
    }
    const float inv = 1.f / (1e-9f + colsum);
    #pragma unroll 4
    for (int k = 0; k < KEXT; ++k)
      ldsB[nn * BSTR + k] = (bf16)(ldsL[nn * LSTR + k] * inv);
  }
  __syncthreads();

  // z = W2(128x64) @ attn(64x128); fuse BN + residual + relu on store.
  {
    const int nn = wave * 16 + (lane & 15);
    const int ng = n0 + nn;
    for (int mt = 0; mt < 8; ++mt) {
      v8f acc = {};
      #pragma unroll
      for (int ks = 0; ks < 2; ++ks) {
        v16bf a = *(const v16bf*)(pA2 + (((mt * 2 + ks) * 32 + lane) * 16));
        v16bf b = *(const v16bf*)(ldsB + nn * BSTR + ks * 32 + halfsel * 16);
        acc = __builtin_amdgcn_wmma_f32_16x16x32_bf16(false, a, false, b,
                                                      (short)0, acc, false, false);
      }
      const int cbase = mt * 16 + halfsel * 8;
      #pragma unroll
      for (int r = 0; r < 8; ++r) {
        const int c = cbase + r;
        float v = acc[r] * sscale[c] + sshift[c] + xb[(size_t)c * NTOT + ng];
        ob[(size_t)c * NTOT + ng] = fmaxf(v, 0.f);
      }
    }
  }
}

// ---------------------------------------------------------------------------
extern "C" void kernel_launch(void* const* d_in, const int* in_sizes, int n_in,
                              void* d_out, int out_size, void* d_ws, size_t ws_size,
                              hipStream_t stream)
{
  (void)in_sizes; (void)n_in; (void)out_size; (void)ws_size;
  const float* x       = (const float*)d_in[0];
  const float* conv1_w = (const float*)d_in[1];
  const float* conv1_b = (const float*)d_in[2];
  const float* lin0_w  = (const float*)d_in[3];
  const float* lin1_w  = (const float*)d_in[4];
  const float* conv2_w = (const float*)d_in[5];
  const float* bn_g    = (const float*)d_in[6];
  const float* bn_b    = (const float*)d_in[7];
  const float* bn_m    = (const float*)d_in[8];
  const float* bn_v    = (const float*)d_in[9];
  float* out = (float*)d_out;

  char* ws = (char*)d_ws;
  bf16*  pA1      = (bf16*)(ws);            // 4*4*32*16 bf16 = 16384 B
  bf16*  pA2      = (bf16*)(ws + 16384);    // 8*2*32*16 bf16 = 16384 B
  float* b1       = (float*)(ws + 32768);   // 64 f32
  float* scale    = (float*)(ws + 33024);   // 128 f32
  float* shift    = (float*)(ws + 33536);   // 128 f32
  float* rowmax   = (float*)(ws + 34048);   // 128 f32
  float* rowinv   = (float*)(ws + 34560);   // 128 f32
  float* partials = (float*)(ws + 36864);   // 2*1024*64*2 f32 = 1 MiB

  ea_prep<<<1, 256, 0, stream>>>(conv1_w, conv1_b, lin0_w, lin1_w, conv2_w,
                                 bn_g, bn_b, bn_m, bn_v,
                                 pA1, pA2, b1, scale, shift);
  dim3 grid(NBLK, 2);
  ea_pass1<<<grid, 256, 0, stream>>>(x, pA1, b1, partials);
  ea_pass2<<<1, 128, 0, stream>>>(partials, rowmax, rowinv);
  ea_pass3<<<grid, 256, 0, stream>>>(x, pA1, pA2, b1, scale, shift,
                                     rowmax, rowinv, out);
}
